// TransformerFusion_44719199486151
// MI455X (gfx1250) — matrix operand order repaired and checked by reading, compiled, not run on hardware
//
#include <hip/hip_runtime.h>
#include <hip/hip_bf16.h>
#include <math.h>

// ---------------------------------------------------------------------------
// Types for CDNA5 WMMA (wave32)
// ---------------------------------------------------------------------------
typedef __attribute__((ext_vector_type(16))) __bf16        v16bf;
typedef __attribute__((ext_vector_type(8)))  float         v8f;
typedef __attribute__((ext_vector_type(4)))  unsigned int  uv4;

union FragBf { uv4 u[2]; v16bf v; };

__device__ __forceinline__ unsigned short f2bf(float f) {
  unsigned int u = __builtin_bit_cast(unsigned int, f);
  u += 0x7FFFu + ((u >> 16) & 1u);            // round-to-nearest-even
  return (unsigned short)(u >> 16);
}
__device__ __forceinline__ float bf2f(unsigned short h) {
  unsigned int u = ((unsigned int)h) << 16;
  return __builtin_bit_cast(float, u);
}

// gfx1250 async memory->LDS copy (VGLOBAL op 98, ASYNCcnt-tracked).
// GVS mode: mem_addr = SADDR(64) + VADDR(i32 byte off); VDST = LDS byte offset.
__device__ __forceinline__ void async_ld_b128(unsigned lds_off,
                                              unsigned long long base,
                                              unsigned byte_off) {
  asm volatile("global_load_async_to_lds_b128 %0, %1, %2"
               :: "v"(lds_off), "v"(byte_off), "s"(base)
               : "memory");
}
__device__ __forceinline__ void wait_async_le(int keep) {
  if (keep == 4) asm volatile("s_wait_asynccnt 0x4" ::: "memory");
  else           asm volatile("s_wait_asynccnt 0x0" ::: "memory");
}

// Problem constants
#define BATCH   8192
#define DMODEL  1024
#define NHEAD   16
#define M3      (BATCH * 3)     // 24576 token rows

// ---------------------------------------------------------------------------
// Weight convert + transpose: W[K,N] f32 -> Wt[N,K] bf16
// ---------------------------------------------------------------------------
__global__ void convT_kernel(const float* __restrict__ W,
                             unsigned short* __restrict__ Wt, int K, int N) {
  long long i = (long long)blockIdx.x * 256 + threadIdx.x;
  if (i >= (long long)K * N) return;
  int k = (int)(i / N), n = (int)(i % N);
  Wt[(size_t)n * K + k] = f2bf(W[i]);
}

// ---------------------------------------------------------------------------
// Modality encoder: seq = relu(LN(sig * w + b) * g + bt), one wave per row
// ---------------------------------------------------------------------------
__global__ __launch_bounds__(256)
void enc_kernel(const float* __restrict__ cs, const float* __restrict__ ss,
                const float* __restrict__ ps,
                const float* __restrict__ ew, const float* __restrict__ eb,
                const float* __restrict__ eg, const float* __restrict__ ebt,
                unsigned short* __restrict__ seq) {
  int gtid = blockIdx.x * 256 + threadIdx.x;
  int row  = gtid >> 5;                       // 0..24575
  int lane = gtid & 31;
  int b = row / 3, mod = row - b * 3;
  float sig = (mod == 0) ? cs[b] : ((mod == 1) ? ss[b] : ps[b]);
  const float* w  = ew  + mod * DMODEL;
  const float* bb = eb  + mod * DMODEL;
  const float* g  = eg  + mod * DMODEL;
  const float* bt = ebt + mod * DMODEL;
  float vals[32]; float s = 0.f, sq = 0.f;
#pragma unroll
  for (int j = 0; j < 32; j++) {
    int d = lane + j * 32;
    float v = fmaf(sig, w[d], bb[d]);
    vals[j] = v; s += v; sq += v * v;
  }
#pragma unroll
  for (int off = 16; off > 0; off >>= 1) {
    s  += __shfl_xor(s,  off, 32);
    sq += __shfl_xor(sq, off, 32);
  }
  float mean = s * (1.f / DMODEL);
  float var  = sq * (1.f / DMODEL) - mean * mean;
  float rstd = rsqrtf(var + 1e-5f);
  unsigned short* orow = seq + (size_t)row * DMODEL;
#pragma unroll
  for (int j = 0; j < 32; j++) {
    int d = lane + j * 32;
    float v = fmaf((vals[j] - mean) * rstd, g[d], bt[d]);
    orow[d] = f2bf(v > 0.f ? v : 0.f);
  }
}

// ---------------------------------------------------------------------------
// Tiled bf16 WMMA GEMM:  C[M,N] = A[M,K] * Bt[N,K]^T + bias, + epilogue
//   MODE 0: bf16 out (QKV)
//   MODE 1: + residual(bf16) -> f32 out AND bf16 out (out-proj)
//   MODE 2: ReLU -> bf16 out (FFN1, fusion1)
//   MODE 3: f32 out (FFN2 pre-LN)
// Block 256 threads (8 waves). Tile 128x128, K-step 32. Wave tile 32x64.
// Double-buffered LDS fed by global_load_async_to_lds_b128 (ASYNCcnt).
// ---------------------------------------------------------------------------
#define BM 128
#define BN 128
#define BK 32
#define LDT 40   // padded LDS row stride (elements); 80B rows, 16B-aligned chunks

template <int MODE>
__global__ __launch_bounds__(256)
void gemm_bf16(const unsigned short* __restrict__ A,    // [M,K] bf16
               const unsigned short* __restrict__ Bt,   // [N,K] bf16 (transposed W)
               const float* __restrict__ bias,          // [N]
               const unsigned short* __restrict__ resBf,// [M,N] bf16 residual (MODE 1)
               unsigned short* __restrict__ outBf,      // [M,N] bf16
               float* __restrict__ outF,                // [M,N] f32
               int M, int N, int K) {
  __shared__ __align__(16) unsigned short sA[2][BM * LDT];
  __shared__ __align__(16) unsigned short sB[2][BN * LDT];

  const int tid  = threadIdx.x;
  const int lane = tid & 31;
  const int wave = tid >> 5;
  const int wm = (wave & 3) * 32;   // wave M origin within block tile
  const int wn = (wave >> 2) * 64;  // wave N origin within block tile
  const int m0 = blockIdx.y * BM;
  const int n0 = blockIdx.x * BN;

  // per-thread async-copy assignment: chunk c covers (row c>>2, 8 elems at (c&3)*8)
  unsigned gOffA[2], gOffB[2];            // global byte offsets (k0=0)
  unsigned ldsA[2][2], ldsB[2][2];        // LDS byte offsets per buffer
#pragma unroll
  for (int i = 0; i < 2; i++) {
    int c = tid + i * 256;
    int r = c >> 2, col = (c & 3) * 8;
    gOffA[i] = (unsigned)(((size_t)(m0 + r) * K + col) * 2);
    gOffB[i] = (unsigned)(((size_t)(n0 + r) * K + col) * 2);
#pragma unroll
    for (int bfi = 0; bfi < 2; bfi++) {
      ldsA[bfi][i] = (unsigned)(uintptr_t)&sA[bfi][r * LDT + col];
      ldsB[bfi][i] = (unsigned)(uintptr_t)&sB[bfi][r * LDT + col];
    }
  }
  const unsigned long long baseA  = (unsigned long long)(uintptr_t)A;
  const unsigned long long baseBt = (unsigned long long)(uintptr_t)Bt;

  v8f acc[2][4];
#pragma unroll
  for (int mi = 0; mi < 2; mi++)
#pragma unroll
    for (int ni = 0; ni < 4; ni++)
#pragma unroll
      for (int i = 0; i < 8; i++) acc[mi][ni][i] = 0.f;

  // prologue: stage tile 0 into buffer 0
#pragma unroll
  for (int i = 0; i < 2; i++) {
    async_ld_b128(ldsA[0][i], baseA,  gOffA[i]);
    async_ld_b128(ldsB[0][i], baseBt, gOffB[i]);
  }

  const int nk = K / BK;
  for (int t = 0; t < nk; t++) {
    const int cur = t & 1;
    if (t + 1 < nk) {
      const unsigned kb = (unsigned)((t + 1) * BK * 2);
#pragma unroll
      for (int i = 0; i < 2; i++) {
        async_ld_b128(ldsA[cur ^ 1][i], baseA,  gOffA[i] + kb);
        async_ld_b128(ldsB[cur ^ 1][i], baseBt, gOffB[i] + kb);
      }
      wait_async_le(4);     // drain tile t's 4 copies (in-order completion)
    } else {
      wait_async_le(0);
    }
    __syncthreads();        // all threads' tile-t copies now visible in LDS

    // ---- A fragments: lanes 0-15 row M=lane K=kb..kb+7 (+16..23 in hi) ----
    FragBf a[2];
#pragma unroll
    for (int mi = 0; mi < 2; mi++) {
      int row = wm + mi * 16 + (lane & 15);
      int kb  = (lane >> 4) * 8;
      a[mi].u[0] = *reinterpret_cast<const uv4*>(&sA[cur][row * LDT + kb]);
      a[mi].u[1] = *reinterpret_cast<const uv4*>(&sA[cur][row * LDT + kb + 16]);
    }
    // ---- B fragments: lane n holds column N=n&15, K=kb..kb+7 (+16..23), kb=(lane>>4)*8 ----
#pragma unroll
    for (int ni = 0; ni < 4; ni++) {
      int col = wn + ni * 16 + (lane & 15);
      int kb  = (lane >> 4) * 8;
      FragBf b;
      b.u[0] = *reinterpret_cast<const uv4*>(&sB[cur][col * LDT + kb]);
      b.u[1] = *reinterpret_cast<const uv4*>(&sB[cur][col * LDT + kb + 16]);
#pragma unroll
      for (int mi = 0; mi < 2; mi++) {
        acc[mi][ni] = __builtin_amdgcn_wmma_f32_16x16x32_bf16(
            false, a[mi].v, false, b.v, (short)0, acc[mi][ni], false, false);
      }
    }
    __syncthreads();        // protect buffer cur before tile t+2's copies land
  }

  // ---- epilogue; C layout: VGPR i -> row (lane<16 ? i : 8+i), col lane&15 ----
#pragma unroll
  for (int mi = 0; mi < 2; mi++) {
#pragma unroll
    for (int ni = 0; ni < 4; ni++) {
#pragma unroll
      for (int i = 0; i < 8; i++) {
        int row = m0 + wm + mi * 16 + ((lane < 16) ? i : 8 + i);
        int col = n0 + wn + ni * 16 + (lane & 15);
        float v = acc[mi][ni][i] + bias[col];
        size_t o = (size_t)row * N + col;
        if constexpr (MODE == 0) {
          outBf[o] = f2bf(v);
        } else if constexpr (MODE == 1) {
          v += bf2f(resBf[o]);
          outF[o]  = v;
          outBf[o] = f2bf(v);
        } else if constexpr (MODE == 2) {
          v = v > 0.f ? v : 0.f;
          outBf[o] = f2bf(v);
        } else {
          outF[o] = v;
        }
      }
    }
  }
}

// ---------------------------------------------------------------------------
// Attention over 3 tokens, one thread per (batch, head)
// qkv: [M3, 3072] bf16 (q | k | v), ctx: [M3, 1024] bf16
// ---------------------------------------------------------------------------
__global__ __launch_bounds__(256)
void attn_kernel(const unsigned short* __restrict__ qkv,
                 unsigned short* __restrict__ ctx) {
  int gid = blockIdx.x * 256 + threadIdx.x;   // b*16 + h
  int b = gid >> 4, h = gid & 15;
  const size_t RS = 3 * DMODEL;               // 3072
  const unsigned short* base = qkv + (size_t)b * 3 * RS + h * 64;

  float sc[3][3];
#pragma unroll
  for (int t = 0; t < 3; t++)
#pragma unroll
    for (int s = 0; s < 3; s++) sc[t][s] = 0.f;

  for (int d = 0; d < 64; d++) {
    float q[3], k[3];
#pragma unroll
    for (int t = 0; t < 3; t++) {
      q[t] = bf2f(base[t * RS + d]);
      k[t] = bf2f(base[t * RS + DMODEL + d]);
    }
#pragma unroll
    for (int t = 0; t < 3; t++)
#pragma unroll
      for (int s = 0; s < 3; s++) sc[t][s] = fmaf(q[t], k[s], sc[t][s]);
  }

  float at[3][3];
#pragma unroll
  for (int t = 0; t < 3; t++) {
    float x0 = sc[t][0] * 0.125f, x1 = sc[t][1] * 0.125f, x2 = sc[t][2] * 0.125f;
    float m = fmaxf(x0, fmaxf(x1, x2));
    float e0 = __expf(x0 - m), e1 = __expf(x1 - m), e2 = __expf(x2 - m);
    float inv = 1.f / (e0 + e1 + e2);
    at[t][0] = e0 * inv; at[t][1] = e1 * inv; at[t][2] = e2 * inv;
  }

  for (int d = 0; d < 64; d++) {
    float v0 = bf2f(base[0 * RS + 2 * DMODEL + d]);
    float v1 = bf2f(base[1 * RS + 2 * DMODEL + d]);
    float v2 = bf2f(base[2 * RS + 2 * DMODEL + d]);
#pragma unroll
    for (int t = 0; t < 3; t++) {
      float o = at[t][0] * v0 + at[t][1] * v1 + at[t][2] * v2;
      ctx[(size_t)(b * 3 + t) * DMODEL + h * 64 + d] = f2bf(o);
    }
  }
}

// ---------------------------------------------------------------------------
// LayerNorm(x)*g+bt + residual, in-place on x, one wave per row
// ---------------------------------------------------------------------------
__global__ __launch_bounds__(256)
void ln_res_kernel(float* __restrict__ x, const float* __restrict__ g,
                   const float* __restrict__ bt, const float* __restrict__ res) {
  int gtid = blockIdx.x * 256 + threadIdx.x;
  int row  = gtid >> 5;
  int lane = gtid & 31;
  float* r = x + (size_t)row * DMODEL;
  const float* rr = res + (size_t)row * DMODEL;
  float vals[32]; float s = 0.f, sq = 0.f;
#pragma unroll
  for (int j = 0; j < 32; j++) {
    int d = lane + j * 32;
    float v = r[d];
    vals[j] = v; s += v; sq += v * v;
  }
#pragma unroll
  for (int off = 16; off > 0; off >>= 1) {
    s  += __shfl_xor(s,  off, 32);
    sq += __shfl_xor(sq, off, 32);
  }
  float mean = s * (1.f / DMODEL);
  float var  = sq * (1.f / DMODEL) - mean * mean;
  float rstd = rsqrtf(var + 1e-5f);
#pragma unroll
  for (int j = 0; j < 32; j++) {
    int d = lane + j * 32;
    r[d] = fmaf((vals[j] - mean) * rstd, g[d], bt[d]) + rr[d];
  }
}

// ---------------------------------------------------------------------------
// Mean pool over 3 tokens -> bf16 pooled [B, D]
// ---------------------------------------------------------------------------
__global__ __launch_bounds__(256)
void pool_kernel(const float* __restrict__ out, unsigned short* __restrict__ pooled) {
  int i = blockIdx.x * 256 + threadIdx.x;     // b*1024 + d
  int b = i >> 10, d = i & 1023;
  size_t r0 = (size_t)(b * 3) * DMODEL + d;
  float v = (out[r0] + out[r0 + DMODEL] + out[r0 + 2 * DMODEL]) * (1.f / 3.f);
  pooled[i] = f2bf(v);
}

// ---------------------------------------------------------------------------
// Fusion head tail: out[b] = tanh(dot(fh[b,:512], wf2) + bf2); one wave per row
// ---------------------------------------------------------------------------
__global__ __launch_bounds__(256)
void head_kernel(const unsigned short* __restrict__ fh, const float* __restrict__ wf2,
                 const float* __restrict__ bf2v, float* __restrict__ out) {
  int gtid = blockIdx.x * 256 + threadIdx.x;
  int b    = gtid >> 5;
  int lane = gtid & 31;
  float s = 0.f;
#pragma unroll
  for (int j = 0; j < 16; j++) {
    int d = lane + j * 32;
    s = fmaf(bf2f(fh[(size_t)b * 512 + d]), wf2[d], s);
  }
#pragma unroll
  for (int off = 16; off > 0; off >>= 1) s += __shfl_xor(s, off, 32);
  if (lane == 0) out[b] = tanhf(s + bf2v[0]);
}

// ---------------------------------------------------------------------------
// Launch
// ---------------------------------------------------------------------------
extern "C" void kernel_launch(void* const* d_in, const int* in_sizes, int n_in,
                              void* d_out, int out_size, void* d_ws, size_t ws_size,
                              hipStream_t stream) {
  (void)in_sizes; (void)n_in; (void)out_size; (void)ws_size;
  const float* cs   = (const float*)d_in[0];
  const float* ss   = (const float*)d_in[1];
  const float* ps   = (const float*)d_in[2];
  const float* ew   = (const float*)d_in[3];
  const float* eb   = (const float*)d_in[4];
  const float* eg   = (const float*)d_in[5];
  const float* ebt  = (const float*)d_in[6];
  const float* wqkv = (const float*)d_in[7];
  const float* bqkv = (const float*)d_in[8];
  const float* wo   = (const float*)d_in[9];
  const float* bo   = (const float*)d_in[10];
  const float* w1   = (const float*)d_in[11];
  const float* b1   = (const float*)d_in[12];
  const float* w2   = (const float*)d_in[13];
  const float* b2   = (const float*)d_in[14];
  const float* g2   = (const float*)d_in[15];
  const float* bt2  = (const float*)d_in[16];
  const float* wf1  = (const float*)d_in[17];
  const float* bf1  = (const float*)d_in[18];
  const float* wf2  = (const float*)d_in[19];
  const float* bf2v = (const float*)d_in[20];

  char* ws = (char*)d_ws;
  // -------- workspace layout (bytes), with lifetime-based overlays --------
  constexpr size_t OFF_SEQ   = 0;                          // 24576*1024*2 = 50.3MB bf16
  constexpr size_t OFF_WQKVT = OFF_SEQ   + (size_t)M3 * DMODEL * 2;          // 6.3MB
  constexpr size_t OFF_WOT   = OFF_WQKVT + (size_t)3 * DMODEL * DMODEL * 2;  // 2MB
  constexpr size_t OFF_W1T   = OFF_WOT   + (size_t)DMODEL * DMODEL * 2;      // 8.4MB
  constexpr size_t OFF_W2T   = OFF_W1T   + (size_t)4 * DMODEL * DMODEL * 2;  // 8.4MB
  constexpr size_t OFF_WF1T  = OFF_W2T   + (size_t)4 * DMODEL * DMODEL * 2;  // 1MB
  constexpr size_t OFF_R     = OFF_WF1T  + (size_t)512 * DMODEL * 2;  // overlay region, 201MB
  constexpr size_t OFF_QKV   = OFF_R;                                  // phase A: qkv 151MB
  constexpr size_t OFF_CTX   = OFF_R + (size_t)M3 * 3 * DMODEL * 2;    // phase A: ctx  50MB
  constexpr size_t OFF_H     = OFF_R;                                  // phase B: h   201MB
  constexpr size_t OFF_POOL  = OFF_R;                                  // phase C: pooled 17MB
  constexpr size_t OFF_FH    = OFF_R + (size_t)BATCH * DMODEL * 2;     // phase C: fh  8.4MB
  constexpr size_t OFF_ATTNF = OFF_R + (size_t)M3 * 4 * DMODEL * 2;    // 100.7MB f32
  constexpr size_t OFF_ATTNB = OFF_ATTNF + (size_t)M3 * DMODEL * 4;    // 50.3MB bf16
  constexpr size_t OFF_FF    = OFF_ATTNB + (size_t)M3 * DMODEL * 2;    // 100.7MB f32

  unsigned short* seqB   = (unsigned short*)(ws + OFF_SEQ);
  unsigned short* wqkvT  = (unsigned short*)(ws + OFF_WQKVT);
  unsigned short* woT    = (unsigned short*)(ws + OFF_WOT);
  unsigned short* w1T    = (unsigned short*)(ws + OFF_W1T);
  unsigned short* w2T    = (unsigned short*)(ws + OFF_W2T);
  unsigned short* wf1T   = (unsigned short*)(ws + OFF_WF1T);
  unsigned short* qkvB   = (unsigned short*)(ws + OFF_QKV);
  unsigned short* ctxB   = (unsigned short*)(ws + OFF_CTX);
  unsigned short* hB     = (unsigned short*)(ws + OFF_H);
  unsigned short* pooledB= (unsigned short*)(ws + OFF_POOL);
  unsigned short* fhB    = (unsigned short*)(ws + OFF_FH);
  float*          attnF  = (float*)(ws + OFF_ATTNF);
  unsigned short* attnB  = (unsigned short*)(ws + OFF_ATTNB);
  float*          ffF    = (float*)(ws + OFF_FF);

  // -------- weight convert+transpose (f32 [K,N] -> bf16 [N,K]) --------
  auto launchT = [&](const float* W, unsigned short* Wt, int K, int N) {
    long long total = (long long)K * N;
    convT_kernel<<<(unsigned)((total + 255) / 256), 256, 0, stream>>>(W, Wt, K, N);
  };
  launchT(wqkv, wqkvT, DMODEL, 3 * DMODEL);
  launchT(wo,   woT,   DMODEL, DMODEL);
  launchT(w1,   w1T,   DMODEL, 4 * DMODEL);
  launchT(w2,   w2T,   4 * DMODEL, DMODEL);
  launchT(wf1,  wf1T,  DMODEL, 512);

  // -------- encoders --------
  enc_kernel<<<M3 * 32 / 256, 256, 0, stream>>>(cs, ss, ps, ew, eb, eg, ebt, seqB);

  dim3 blk(256);
  // -------- QKV projection --------
  gemm_bf16<0><<<dim3(3 * DMODEL / BN, M3 / BM), blk, 0, stream>>>(
      seqB, wqkvT, bqkv, nullptr, qkvB, nullptr, M3, 3 * DMODEL, DMODEL);
  // -------- attention (3x3 per head) --------
  attn_kernel<<<BATCH * NHEAD / 256, 256, 0, stream>>>(qkvB, ctxB);
  // -------- out-proj + residual(seq) --------
  gemm_bf16<1><<<dim3(DMODEL / BN, M3 / BM), blk, 0, stream>>>(
      ctxB, woT, bo, seqB, attnB, attnF, M3, DMODEL, DMODEL);
  // -------- FFN up (ReLU) --------
  gemm_bf16<2><<<dim3(4 * DMODEL / BN, M3 / BM), blk, 0, stream>>>(
      attnB, w1T, b1, nullptr, hB, nullptr, M3, 4 * DMODEL, DMODEL);
  // -------- FFN down (pre-LN f32) --------
  gemm_bf16<3><<<dim3(DMODEL / BN, M3 / BM), blk, 0, stream>>>(
      hB, w2T, b2, nullptr, nullptr, ffF, M3, DMODEL, 4 * DMODEL);
  // -------- LN + residual (in-place on ffF) --------
  ln_res_kernel<<<M3 * 32 / 256, 256, 0, stream>>>(ffF, g2, bt2, attnF);
  // -------- mean pool over 3 tokens --------
  pool_kernel<<<BATCH * DMODEL / 256, 256, 0, stream>>>(ffF, pooledB);
  // -------- fusion head layer 1 (ReLU) --------
  gemm_bf16<2><<<dim3(512 / BN, BATCH / BM), blk, 0, stream>>>(
      pooledB, wf1T, bf1, nullptr, fhB, nullptr, BATCH, 512, DMODEL);
  // -------- fusion head layer 2 + tanh --------
  head_kernel<<<BATCH * 32 / 256, 256, 0, stream>>>(fhB, wf2, bf2v, (float*)d_out);
}
